// GCNPredictionNet2_46084999086840
// MI455X (gfx1250) — matrix operand order repaired and checked by reading, not compiled, not run on hardware
//
#include <hip/hip_runtime.h>
#include <math.h>

#define N_NODES 100000
#define N_EDGES 6400000

typedef __attribute__((ext_vector_type(16))) _Float16 v16h;
typedef __attribute__((ext_vector_type(8)))  float    v8f;

__device__ __forceinline__ float softplusf(float x) {
  // numerically stable log(1+exp(x))
  return x > 0.0f ? x + log1pf(expf(-x)) : log1pf(expf(x));
}

// ---------------------------------------------------------------------------
// Zero workspace accumulators (sums1, cnt, sums2)
// ---------------------------------------------------------------------------
__global__ void k_zero(float* __restrict__ p, long n) {
  long i = (long)blockIdx.x * blockDim.x + threadIdx.x;
  long stride = (long)gridDim.x * blockDim.x;
  for (; i < n; i += stride) p[i] = 0.0f;
}

// ---------------------------------------------------------------------------
// Layer 1 scatter: one wave per edge-iteration, lane = feature (32 features).
// Grid-stride persistent waves + software prefetch of the next random row.
// sums1[dst][f] += x[src][f];  cnt[dst] += 1
// ---------------------------------------------------------------------------
__global__ void k_scatter1(const float* __restrict__ x,
                           const int* __restrict__ src,
                           const int* __restrict__ dst,
                           float* __restrict__ sums1,
                           float* __restrict__ cnt) {
  const int f = threadIdx.x & 31;
  const long w  = ((long)blockIdx.x * blockDim.x + threadIdx.x) >> 5;
  const long nw = ((long)gridDim.x * blockDim.x) >> 5;
  for (long e = w; e < N_EDGES; e += nw) {
    int s = src[e];
    int d = dst[e];
    // prefetch the next iteration's gathered row (random address known early)
    long en = e + nw;
    if (en < N_EDGES) {
      int sn = src[en];
      __builtin_prefetch(&x[(long)sn * 32 + f], 0, 0);
    }
    float v = x[(long)s * 32 + f];
    atomicAdd(&sums1[(long)d * 32 + f], v);
    if (f == 0) atomicAdd(&cnt[d], 1.0f);
  }
}

// ---------------------------------------------------------------------------
// Layer 2 scatter: 16 lanes per edge (h1 padded to 16, col 15 == 0).
// Grid-stride + prefetch, two edges per wave per iteration.
// ---------------------------------------------------------------------------
__global__ void k_scatter2(const float* __restrict__ h1,
                           const int* __restrict__ src,
                           const int* __restrict__ dst,
                           float* __restrict__ sums2) {
  const int f = threadIdx.x & 15;
  const long slot = ((long)blockIdx.x * blockDim.x + threadIdx.x) >> 4;
  const long ns   = ((long)gridDim.x * blockDim.x) >> 4;
  for (long e = slot; e < N_EDGES; e += ns) {
    int s = src[e];
    int d = dst[e];
    long en = e + ns;
    if (en < N_EDGES) {
      int sn = src[en];
      __builtin_prefetch(&h1[(long)sn * 16 + f], 0, 0);
    }
    if (f < 15) {
      atomicAdd(&sums2[(long)d * 16 + f], h1[(long)s * 16 + f]);
    }
  }
}

// ---------------------------------------------------------------------------
// Node kernel 1: h1 = softplus(mean1 @ W1_rel + b1 + x @ W1_root)
// One 16x16 output tile per wave, two chained v_wmma_f32_16x16x32_f16.
// Block = 128 threads (4 waves) -> 64 node rows per block.
// Tile staging uses B128 loads/stores; mean uses one rcp per row.
// ---------------------------------------------------------------------------
__global__ void __launch_bounds__(128)
k_node1(const float* __restrict__ x,
        const float* __restrict__ sums1,
        const float* __restrict__ cnt,
        const float* __restrict__ W1_rel,   // [32][15]
        const float* __restrict__ b1_rel,   // [15]
        const float* __restrict__ W1_root,  // [32][15]
        float* __restrict__ h1,             // [N][16], col 15 = 0
        int n_nodes) {
  __shared__ float ldsM[4][16][32];
  __shared__ float ldsX[4][16][32];
  const int lane = threadIdx.x & 31;
  const int wave = threadIdx.x >> 5;
  const int r0 = blockIdx.x * 64 + wave * 16;

  const float4* s1v = (const float4*)sums1;  // [N][8] float4
  const float4* xv4 = (const float4*)x;      // [N][8] float4

  // Cooperative, coalesced B128 tile load: 16 rows x 8 float4.
#pragma unroll
  for (int i = 0; i < 4; ++i) {
    int idx = i * 32 + lane;   // float4 index within 16x8 tile
    int r = idx >> 3;
    int c4 = idx & 7;
    int row = r0 + r;
    float4 mv = make_float4(0.f, 0.f, 0.f, 0.f);
    float4 xv = make_float4(0.f, 0.f, 0.f, 0.f);
    if (row < n_nodes) {
      float rc = 1.0f / fmaxf(cnt[row], 1.0f);
      float4 sv = s1v[(long)row * 8 + c4];
      mv = make_float4(sv.x * rc, sv.y * rc, sv.z * rc, sv.w * rc);
      xv = xv4[(long)row * 8 + c4];
    }
    *(float4*)&ldsM[wave][r][c4 * 4] = mv;
    *(float4*)&ldsX[wave][r][c4 * 4] = xv;
  }
  __syncthreads();

  const int m  = lane & 15;   // A row / C column index base
  const int hf = lane >> 4;   // lane half

  v16h aM, aX, bRel, bRoot;
#pragma unroll
  for (int v = 0; v < 8; ++v) {
    // A-matrix 16x32 f16 layout (ISA 7.12.2): lanes0-15 K=0-7 (V0-3), K=16-23 (V4-7);
    // lanes16-31 K=8-15 and K=24-31.
    int k0 = (v < 4) ? (hf * 8 + 2 * v) : (16 + hf * 8 + 2 * (v - 4));
    aM[2 * v]     = (_Float16)ldsM[wave][m][k0];
    aM[2 * v + 1] = (_Float16)ldsM[wave][m][k0 + 1];
    aX[2 * v]     = (_Float16)ldsX[wave][m][k0];
    aX[2 * v + 1] = (_Float16)ldsX[wave][m][k0 + 1];
    // B-matrix 32x16 f16 layout: same K order as A (lanes0-15 K=0-7,16-23; lanes16-31 K=8-15,24-31).
    int kb = k0;
    float w0r = (m < 15) ? W1_rel[kb * 15 + m]        : 0.0f;
    float w1r = (m < 15) ? W1_rel[(kb + 1) * 15 + m]  : 0.0f;
    float w0t = (m < 15) ? W1_root[kb * 15 + m]       : 0.0f;
    float w1t = (m < 15) ? W1_root[(kb + 1) * 15 + m] : 0.0f;
    bRel[2 * v]      = (_Float16)w0r;
    bRel[2 * v + 1]  = (_Float16)w1r;
    bRoot[2 * v]     = (_Float16)w0t;
    bRoot[2 * v + 1] = (_Float16)w1t;
  }

  v8f c = {};
  c = __builtin_amdgcn_wmma_f32_16x16x32_f16(false, aM, false, bRel,
                                             (short)0, c, false, false);
  c = __builtin_amdgcn_wmma_f32_16x16x32_f16(false, aX, false, bRoot,
                                             (short)0, c, false, false);

  const float bias = (m < 15) ? b1_rel[m] : 0.0f;
#pragma unroll
  for (int v = 0; v < 8; ++v) {
    int row = r0 + v + 8 * hf;  // C/D layout: M = v + 8*(lane/16), N = lane%16
    if (row < n_nodes) {
      float val = (m < 15) ? softplusf(c[v] + bias) : 0.0f;
      h1[(long)row * 16 + m] = val;
    }
  }
}

// ---------------------------------------------------------------------------
// Node kernel 2 (fused): h2 = softplus([mean2 | h1] @ [W2_rel ; W2_root] + b2)
// via ONE wmma per tile, then per-row FC: softplus(h2@W_fc1+b1)@W_fc2+b2 -> out.
// ---------------------------------------------------------------------------
__global__ void __launch_bounds__(128)
k_node2(const float* __restrict__ h1,       // [N][16]
        const float* __restrict__ sums2,    // [N][16]
        const float* __restrict__ cnt,
        const float* __restrict__ W2_rel,   // [15][10]
        const float* __restrict__ b2_rel,   // [10]
        const float* __restrict__ W2_root,  // [15][10]
        const float* __restrict__ W_fc1,    // [10][5]
        const float* __restrict__ b_fc1,    // [5]
        const float* __restrict__ W_fc2,    // [5][1]
        const float* __restrict__ b_fc2,    // [1]
        float* __restrict__ out,            // [N]
        int n_nodes) {
  __shared__ float ldsA[4][16][32];
  __shared__ float ldsH[64][16];
  const int lane = threadIdx.x & 31;
  const int wave = threadIdx.x >> 5;
  const int r0 = blockIdx.x * 64 + wave * 16;

  const float4* s2v = (const float4*)sums2;  // [N][4] float4
  const float4* h1v = (const float4*)h1;     // [N][4] float4

  // A tile: columns 0-15 = mean2, columns 16-31 = h1 (both 16-padded).
  // B128 staged: 16 rows x 8 float4 (c4 0-3 -> mean2, c4 4-7 -> h1).
#pragma unroll
  for (int i = 0; i < 4; ++i) {
    int idx = i * 32 + lane;
    int r = idx >> 3;
    int c4 = idx & 7;
    int row = r0 + r;
    float4 v = make_float4(0.f, 0.f, 0.f, 0.f);
    if (row < n_nodes) {
      if (c4 < 4) {
        float rc = 1.0f / fmaxf(cnt[row], 1.0f);
        float4 sv = s2v[(long)row * 4 + c4];
        v = make_float4(sv.x * rc, sv.y * rc, sv.z * rc, sv.w * rc);
      } else {
        v = h1v[(long)row * 4 + (c4 - 4)];
      }
    }
    *(float4*)&ldsA[wave][r][c4 * 4] = v;
  }
  __syncthreads();

  const int m  = lane & 15;
  const int hf = lane >> 4;

  v16h a, b;
#pragma unroll
  for (int v = 0; v < 8; ++v) {
    int k0 = (v < 4) ? (hf * 8 + 2 * v) : (16 + hf * 8 + 2 * (v - 4));
    a[2 * v]     = (_Float16)ldsA[wave][m][k0];
    a[2 * v + 1] = (_Float16)ldsA[wave][m][k0 + 1];
    int kb = k0;
#pragma unroll
    for (int t = 0; t < 2; ++t) {
      int k = kb + t;
      float w = 0.0f;
      if (m < 10) {
        if (k < 16) {
          if (k < 15) w = W2_rel[k * 10 + m];
        } else {
          int kk = k - 16;
          if (kk < 15) w = W2_root[kk * 10 + m];
        }
      }
      b[2 * v + t] = (_Float16)w;
    }
  }

  v8f c = {};
  c = __builtin_amdgcn_wmma_f32_16x16x32_f16(false, a, false, b,
                                             (short)0, c, false, false);

  const float bias = (m < 10) ? b2_rel[m] : 0.0f;
#pragma unroll
  for (int v = 0; v < 8; ++v) {
    int ml = v + 8 * hf;
    float val = (m < 10) ? softplusf(c[v] + bias) : 0.0f;
    ldsH[wave * 16 + ml][m] = val;
  }
  __syncthreads();

  // Final tiny MLP, one thread per node row.
  int t = threadIdx.x;
  if (t < 64) {
    int row = blockIdx.x * 64 + t;
    if (row < n_nodes) {
      float acc2 = b_fc2[0];
#pragma unroll
      for (int j = 0; j < 5; ++j) {
        float aj = b_fc1[j];
#pragma unroll
        for (int k = 0; k < 10; ++k) aj += ldsH[t][k] * W_fc1[k * 5 + j];
        acc2 += softplusf(aj) * W_fc2[j];
      }
      out[row] = acc2;
    }
  }
}

// ---------------------------------------------------------------------------
// Min reduction (two-stage) + subtract
// ---------------------------------------------------------------------------
__global__ void k_min1(const float* __restrict__ v, float* __restrict__ bmins,
                       int n) {
  __shared__ float s[256];
  float m = 3.402823466e38f;
  for (int i = blockIdx.x * 256 + threadIdx.x; i < n; i += gridDim.x * 256)
    m = fminf(m, v[i]);
  s[threadIdx.x] = m;
  __syncthreads();
  for (int off = 128; off > 0; off >>= 1) {
    if (threadIdx.x < off)
      s[threadIdx.x] = fminf(s[threadIdx.x], s[threadIdx.x + off]);
    __syncthreads();
  }
  if (threadIdx.x == 0) bmins[blockIdx.x] = s[0];
}

__global__ void k_min2(const float* __restrict__ bmins, float* __restrict__ gmin,
                       int nb) {
  __shared__ float s[256];
  float m = 3.402823466e38f;
  for (int i = threadIdx.x; i < nb; i += 256) m = fminf(m, bmins[i]);
  s[threadIdx.x] = m;
  __syncthreads();
  for (int off = 128; off > 0; off >>= 1) {
    if (threadIdx.x < off)
      s[threadIdx.x] = fminf(s[threadIdx.x], s[threadIdx.x + off]);
    __syncthreads();
  }
  if (threadIdx.x == 0) *gmin = s[0];
}

__global__ void k_sub(float* __restrict__ v, const float* __restrict__ gmin,
                      int n) {
  int i = blockIdx.x * blockDim.x + threadIdx.x;
  if (i < n) v[i] -= *gmin;
}

// ---------------------------------------------------------------------------
extern "C" void kernel_launch(void* const* d_in, const int* in_sizes, int n_in,
                              void* d_out, int out_size, void* d_ws,
                              size_t ws_size, hipStream_t stream) {
  const float* x       = (const float*)d_in[0];
  const int*   ei      = (const int*)d_in[1];
  const float* W1_rel  = (const float*)d_in[2];
  const float* b1_rel  = (const float*)d_in[3];
  const float* W1_root = (const float*)d_in[4];
  const float* W2_rel  = (const float*)d_in[5];
  const float* b2_rel  = (const float*)d_in[6];
  const float* W2_root = (const float*)d_in[7];
  const float* W_fc1   = (const float*)d_in[8];
  const float* b_fc1   = (const float*)d_in[9];
  const float* W_fc2   = (const float*)d_in[10];
  const float* b_fc2   = (const float*)d_in[11];
  float* out = (float*)d_out;

  float* ws    = (float*)d_ws;
  float* sums1 = ws;                          // N*32
  float* cnt   = ws + (long)N_NODES * 32;     // N
  float* sums2 = ws + (long)N_NODES * 33;     // N*16
  float* h1    = ws + (long)N_NODES * 49;     // N*16
  float* bmins = ws + (long)N_NODES * 65;     // 256
  float* gmin  = bmins + 256;                 // 1

  const int* src = ei;
  const int* dst = ei + N_EDGES;

  // 1) zero accumulators (sums1, cnt, sums2 = 49*N floats)
  k_zero<<<2048, 256, 0, stream>>>(ws, (long)N_NODES * 49);

  // 2) layer-1 message scatter (persistent waves, lane=feature, prefetch)
  k_scatter1<<<16384, 256, 0, stream>>>(x, src, dst, sums1, cnt);

  // 3) layer-1 node update (WMMA): h1 = softplus(mean1@W1_rel + b1 + x@W1_root)
  k_node1<<<(N_NODES + 63) / 64, 128, 0, stream>>>(x, sums1, cnt, W1_rel,
                                                   b1_rel, W1_root, h1,
                                                   N_NODES);

  // 4) layer-2 message scatter (16 lanes/edge, prefetch)
  k_scatter2<<<16384, 256, 0, stream>>>(h1, src, dst, sums2);

  // 5) layer-2 node update + fused FC head (single WMMA per tile)
  k_node2<<<(N_NODES + 63) / 64, 128, 0, stream>>>(
      h1, sums2, cnt, W2_rel, b2_rel, W2_root, W_fc1, b_fc1, W_fc2, b_fc2, out,
      N_NODES);

  // 6) out -= min(out)
  k_min1<<<256, 256, 0, stream>>>(out, bmins, N_NODES);
  k_min2<<<1, 256, 0, stream>>>(bmins, gmin, 256);
  k_sub<<<(N_NODES + 255) / 256, 256, 0, stream>>>(out, gmin, N_NODES);
}
